// TransformerDQN_20401094656080
// MI455X (gfx1250) — compile-verified, matrix operand order checked by reading, not run on hardware
//
#include <hip/hip_runtime.h>
#include <hip/hip_bf16.h>
#include <math.h>

// ---------------------------------------------------------------------------
// Fused 4-layer transformer forward for MI455X (gfx1250, wave32, WMMA).
// One workgroup (8 wave32s) per board sample; whole sample resident in LDS.
// All matmuls via v_wmma_f32_16x16x32_f16. Weights pre-converted to f16 and
// pre-transposed into d_ws so B-fragments are contiguous global_load_b128.
// ---------------------------------------------------------------------------

typedef __attribute__((ext_vector_type(16))) _Float16 v16h;
typedef __attribute__((ext_vector_type(8)))  _Float16 v8h;
typedef __attribute__((ext_vector_type(8)))  float    v8f;

#define DIM   128
#define NH    4
#define NL    4
#define DFFN  512
#define RR    6
#define CCOLS 7
#define SEQ   43     // 1 + 6*7 real tokens
#define SP    48     // padded rows (3 x 16)
#define KPAD  64     // padded keys (2 x 32 WMMA K-steps)
#define NBUCK 143    // (2*(R-1)+1)*(2*(C-1)+1)
#define DKH   32     // head dim
#define QSCALE 0.17677669529663687f  // 1/sqrt(32)

// LDS layout (bytes)
#define OFF_X    0
#define OFF_A    (OFF_X  + SP*DIM*4)   // ln output (f16), reused for ln2
#define OFF_Q    (OFF_A  + SP*DIM*2)   // q (f16, pre-scaled)
#define OFF_K    (OFF_Q  + SP*DIM*2)   // k (f16)
#define OFF_VT   (OFF_K  + SP*DIM*2)   // v transposed [DIM][KPAD] f16
#define OFF_ATT  (OFF_VT + DIM*KPAD*2) // attention output (f16)
#define OFF_P    (OFF_ATT+ SP*DIM*2)   // probs [SP][KPAD] f16
#define OFF_SC   (OFF_P  + SP*KPAD*2)  // scores f32 [SP][KPAD]; aliased by FFN g16
#define SMEM_BYTES (OFF_SC + SP*KPAD*4)  // = 108544 bytes (~106 KB < 320 KB/WGP)

#define ASSUME_GLOBAL(p)                                              \
  __builtin_assume(!__builtin_amdgcn_is_shared((const void*)(p)));    \
  __builtin_assume(!__builtin_amdgcn_is_private((const void*)(p)))

__device__ __forceinline__ v8f wmma_f16(v16h a, v16h b, v8f c) {
  // D = A(16x32 f16) * B(32x16 f16) + C(16x16 f32)
  return __builtin_amdgcn_wmma_f32_16x16x32_f16(
      /*neg_a=*/false, a, /*neg_b=*/false, b,
      /*c_mod=*/(short)0, c, /*reuse_a=*/false, /*reuse_b=*/false);
}

// ISA 16-bit A layout: lanes 0-15 hold K=0..7 then 16..23; lanes 16-31 hold
// K=8..15 then 24..31 -> two contiguous 16-byte loads per lane.
// LDS variant (base provably in shared memory -> ds_load_b128).
__device__ __forceinline__ v16h load_frag_lds(const _Float16* base,
                                              int row0, int k0, int ld, int lane) {
  int r  = row0 + (lane & 15);
  int ko = k0 + ((lane >> 4) << 3);
  const _Float16* p = base + (size_t)r * ld + ko;
  v8h lo = *(const v8h*)(p);
  v8h hi = *(const v8h*)(p + 16);
  return __builtin_shufflevector(lo, hi,
      0,1,2,3,4,5,6,7,8,9,10,11,12,13,14,15);
}

// Global variant: assume not-LDS/not-private so the backend emits
// global_load_b128 instead of flat_load_b128 (no DScnt coupling).
__device__ __forceinline__ v16h load_frag_g(const _Float16* __restrict__ base,
                                            int row0, int k0, int ld, int lane) {
  ASSUME_GLOBAL(base);
  int r  = row0 + (lane & 15);
  int ko = k0 + ((lane >> 4) << 3);
  const _Float16* p = base + (size_t)r * ld + ko;
  v8h lo = *(const v8h*)(p);
  v8h hi = *(const v8h*)(p + 16);
  return __builtin_shufflevector(lo, hi,
      0,1,2,3,4,5,6,7,8,9,10,11,12,13,14,15);
}

__device__ __forceinline__ float wave_sum(float v) {
  #pragma unroll
  for (int m = 16; m >= 1; m >>= 1) v += __shfl_xor(v, m, 32);
  return v;
}
__device__ __forceinline__ float wave_max(float v) {
  #pragma unroll
  for (int m = 16; m >= 1; m >>= 1) v = fmaxf(v, __shfl_xor(v, m, 32));
  return v;
}

// LayerNorm of X (f32, [SP][DIM]) -> dst (f16), pad rows zeroed.
// One wave per row; 32 lanes x 4 contiguous columns.
__device__ __forceinline__ void layernorm_rows(const float* X,
                                               const float* __restrict__ sc,
                                               const float* __restrict__ bi,
                                               _Float16* dst,
                                               int wave, int lane) {
  ASSUME_GLOBAL(sc);
  ASSUME_GLOBAL(bi);
  for (int r = wave; r < SP; r += 8) {
    float vals[4];
    float sum = 0.f, sum2 = 0.f;
    #pragma unroll
    for (int j = 0; j < 4; ++j) {
      float v = X[r*DIM + lane*4 + j];
      vals[j] = v; sum += v; sum2 += v*v;
    }
    sum  = wave_sum(sum);
    sum2 = wave_sum(sum2);
    float mu   = sum  * (1.f/DIM);
    float var  = sum2 * (1.f/DIM) - mu*mu;
    float rinv = rsqrtf(var + 1e-5f);
    #pragma unroll
    for (int j = 0; j < 4; ++j) {
      float y = (vals[j]-mu)*rinv*sc[lane*4+j] + bi[lane*4+j];
      dst[r*DIM + lane*4 + j] = (r < SEQ) ? (_Float16)y : (_Float16)0.f;
    }
  }
}

// f32 [L][K][N] -> f16 transposed [L][N][K] (weights as WMMA "B" operands)
__global__ void cvt_transpose_kernel(const float* __restrict__ in,
                                     _Float16* __restrict__ out,
                                     int Lc, int Kc, int Nc) {
  int idx = blockIdx.x * blockDim.x + threadIdx.x;
  int total = Lc * Kc * Nc;
  if (idx >= total) return;
  int l = idx / (Kc * Nc);
  int rem = idx - l * (Kc * Nc);
  int k = rem / Nc;
  int n = rem - k * Nc;
  out[((size_t)l * Nc + n) * Kc + k] = (_Float16)in[idx];
}

__global__ __launch_bounds__(256)
void fused_transformer_fwd(
    const int* __restrict__ board, const int* __restrict__ player,
    const int* __restrict__ bucket,
    const float* __restrict__ cell_emb, const float* __restrict__ player_emb,
    const float* __restrict__ cls_token, const float* __restrict__ bias_table,
    const float* __restrict__ ln1_s, const float* __restrict__ ln1_b,
    const float* __restrict__ bq, const float* __restrict__ bk,
    const float* __restrict__ bv, const float* __restrict__ bo,
    const float* __restrict__ ln2_s, const float* __restrict__ ln2_b,
    const float* __restrict__ b1, const float* __restrict__ b2,
    const float* __restrict__ lnf_s, const float* __restrict__ lnf_b,
    const float* __restrict__ Whead, const float* __restrict__ bhead,
    const _Float16* __restrict__ wqT, const _Float16* __restrict__ wkT,
    const _Float16* __restrict__ wvT, const _Float16* __restrict__ woT,
    const _Float16* __restrict__ w1T, const _Float16* __restrict__ w2T,
    float* __restrict__ out_logits, float* __restrict__ out_masked,
    float* __restrict__ out_valid) {

  extern __shared__ char smem[];
  float*    X    = (float*)   (smem + OFF_X);
  _Float16* A16  = (_Float16*)(smem + OFF_A);
  _Float16* Q16  = (_Float16*)(smem + OFF_Q);
  _Float16* K16  = (_Float16*)(smem + OFF_K);
  _Float16* VT   = (_Float16*)(smem + OFF_VT);
  _Float16* ATT  = (_Float16*)(smem + OFF_ATT);
  _Float16* P16  = (_Float16*)(smem + OFF_P);
  float*    SC   = (float*)   (smem + OFF_SC);
  _Float16* G16  = (_Float16*)(smem + OFF_SC);   // FFN intermediate aliases SC

  // Promote all divergent global reads out of the flat path.
  ASSUME_GLOBAL(board);      ASSUME_GLOBAL(bucket);
  ASSUME_GLOBAL(cell_emb);   ASSUME_GLOBAL(player_emb);
  ASSUME_GLOBAL(cls_token);  ASSUME_GLOBAL(bias_table);
  ASSUME_GLOBAL(bq);  ASSUME_GLOBAL(bk);  ASSUME_GLOBAL(bv);
  ASSUME_GLOBAL(bo);  ASSUME_GLOBAL(b1);  ASSUME_GLOBAL(b2);
  ASSUME_GLOBAL(Whead);

  const int b    = blockIdx.x;
  const int tid  = threadIdx.x;
  const int wave = tid >> 5;
  const int lane = tid & 31;

  // Warm the L2 path for layer-0 weights (global_prefetch_b8 on gfx1250).
  __builtin_prefetch(wqT + (size_t)tid * 64, 0, 1);
  __builtin_prefetch(w1T + (size_t)tid * 64, 0, 1);

  // ---- Embedding: x[0]=cls+player_emb, x[1..42]=cell_emb[board], pad=0 ----
  for (int i = tid; i < SP*DIM; i += 256) {
    int s = i >> 7, d = i & 127;
    float v = 0.f;
    if (s == 0)            v = cls_token[d] + player_emb[player[b]*DIM + d];
    else if (s < SEQ)      v = cell_emb[board[b*(RR*CCOLS) + (s-1)]*DIM + d];
    X[i] = v;
  }
  // Zero V^T once so padded keys (48..63) multiply as exact zeros.
  for (int i = tid; i < DIM*KPAD; i += 256) VT[i] = (_Float16)0.f;
  __syncthreads();

  for (int l = 0; l < NL; ++l) {
    const _Float16* WqTl = wqT + (size_t)l*DIM*DIM;
    const _Float16* WkTl = wkT + (size_t)l*DIM*DIM;
    const _Float16* WvTl = wvT + (size_t)l*DIM*DIM;
    const _Float16* WoTl = woT + (size_t)l*DIM*DIM;
    const _Float16* W1Tl = w1T + (size_t)l*DFFN*DIM;
    const _Float16* W2Tl = w2T + (size_t)l*DIM*DFFN;

    // ---- LN1 -> A16 (f16) ----
    layernorm_rows(X, ln1_s + l*DIM, ln1_b + l*DIM, A16, wave, lane);
    __syncthreads();

    // ---- QKV: three 48x128x128 GEMMs with straight-line epilogues --------
    // Q (pre-scaled by 1/sqrt(dk))
    for (int t = wave; t < 24; t += 8) {
      int mt = t >> 3, nt = t & 7;
      v8f c = {0,0,0,0,0,0,0,0};
      #pragma unroll
      for (int kk = 0; kk < 4; ++kk) {
        v16h a  = load_frag_lds(A16,  mt*16, kk*32, DIM, lane);
        v16h bf = load_frag_g  (WqTl, nt*16, kk*32, DIM, lane);
        c = wmma_f16(a, bf, c);
      }
      int col = nt*16 + (lane & 15);
      float bb = bq[l*DIM + col];
      #pragma unroll
      for (int i = 0; i < 8; ++i) {
        int row = mt*16 + ((lane >> 4) << 3) + i;
        float v = (row < SEQ) ? (c[i] + bb) * QSCALE : 0.f;
        Q16[row*DIM + col] = (_Float16)v;
      }
    }
    // K
    for (int t = wave; t < 24; t += 8) {
      int mt = t >> 3, nt = t & 7;
      v8f c = {0,0,0,0,0,0,0,0};
      #pragma unroll
      for (int kk = 0; kk < 4; ++kk) {
        v16h a  = load_frag_lds(A16,  mt*16, kk*32, DIM, lane);
        v16h bf = load_frag_g  (WkTl, nt*16, kk*32, DIM, lane);
        c = wmma_f16(a, bf, c);
      }
      int col = nt*16 + (lane & 15);
      float bb = bk[l*DIM + col];
      #pragma unroll
      for (int i = 0; i < 8; ++i) {
        int row = mt*16 + ((lane >> 4) << 3) + i;
        float v = (row < SEQ) ? c[i] + bb : 0.f;
        K16[row*DIM + col] = (_Float16)v;
      }
    }
    // V (stored transposed: VT[dk][key])
    for (int t = wave; t < 24; t += 8) {
      int mt = t >> 3, nt = t & 7;
      v8f c = {0,0,0,0,0,0,0,0};
      #pragma unroll
      for (int kk = 0; kk < 4; ++kk) {
        v16h a  = load_frag_lds(A16,  mt*16, kk*32, DIM, lane);
        v16h bf = load_frag_g  (WvTl, nt*16, kk*32, DIM, lane);
        c = wmma_f16(a, bf, c);
      }
      int col = nt*16 + (lane & 15);
      float bb = bv[l*DIM + col];
      #pragma unroll
      for (int i = 0; i < 8; ++i) {
        int row = mt*16 + ((lane >> 4) << 3) + i;
        float v = (row < SEQ) ? c[i] + bb : 0.f;
        VT[col*KPAD + row] = (_Float16)v;
      }
    }
    __syncthreads();

    // ---- Attention, one head at a time (DK=32 => single WMMA K-step) ----
    for (int h = 0; h < NH; ++h) {
      // scores = q @ k^T + bias (3x3 tiles)
      for (int t = wave; t < 9; t += 8) {
        int mt = t / 3, nt = t % 3;
        v16h a  = load_frag_lds(Q16, mt*16, h*DKH, DIM, lane);
        v16h bf = load_frag_lds(K16, nt*16, h*DKH, DIM, lane);
        v8f c = {0,0,0,0,0,0,0,0};
        c = wmma_f16(a, bf, c);
        int colb = nt*16 + (lane & 15);
        #pragma unroll
        for (int i = 0; i < 8; ++i) {
          int row = mt*16 + ((lane >> 4) << 3) + i;
          float v;
          if (colb >= SEQ) v = -1e9f;
          else {
            int rq = (row < SEQ) ? row : 0;
            v = c[i] + bias_table[h*NBUCK + bucket[rq*SEQ + colb]];
          }
          SC[row*KPAD + colb] = v;
        }
      }
      __syncthreads();

      // softmax rows (wave32 shuffle reductions) -> P16 (f16, padded keys=0)
      for (int r = wave; r < SP; r += 8) {
        int c0 = lane, c1 = lane + 32;
        float v0 = (c0 < SEQ) ? SC[r*KPAD + c0] : -3.0e38f;
        float v1 = (c1 < SEQ) ? SC[r*KPAD + c1] : -3.0e38f;
        float m  = wave_max(fmaxf(v0, v1));
        float e0 = (c0 < SEQ) ? __expf(v0 - m) : 0.f;
        float e1 = (c1 < SEQ) ? __expf(v1 - m) : 0.f;
        float inv = 1.f / wave_sum(e0 + e1);
        P16[r*KPAD + c0] = (_Float16)(e0 * inv);
        P16[r*KPAD + c1] = (_Float16)(e1 * inv);
      }
      __syncthreads();

      // attn = probs @ v  (3x2 tiles, K=64 padded -> 2 steps)
      const _Float16* VTh = VT + (size_t)h*DKH*KPAD;
      for (int t = wave; t < 6; t += 8) {
        int mt = t >> 1, nt = t & 1;
        v8f c = {0,0,0,0,0,0,0,0};
        #pragma unroll
        for (int kk = 0; kk < 2; ++kk) {
          v16h a  = load_frag_lds(P16, mt*16, kk*32, KPAD, lane);
          v16h bf = load_frag_lds(VTh, nt*16, kk*32, KPAD, lane);
          c = wmma_f16(a, bf, c);
        }
        int col = h*DKH + nt*16 + (lane & 15);
        #pragma unroll
        for (int i = 0; i < 8; ++i) {
          int row = mt*16 + ((lane >> 4) << 3) + i;
          ATT[row*DIM + col] = (row < SEQ) ? (_Float16)c[i] : (_Float16)0.f;
        }
      }
      __syncthreads();
    }

    // ---- x += attn @ Wo + bo ----
    for (int t = wave; t < 24; t += 8) {
      int mt = t >> 3, nt = t & 7;
      v8f c = {0,0,0,0,0,0,0,0};
      #pragma unroll
      for (int kk = 0; kk < 4; ++kk) {
        v16h a  = load_frag_lds(ATT,  mt*16, kk*32, DIM, lane);
        v16h bf = load_frag_g  (WoTl, nt*16, kk*32, DIM, lane);
        c = wmma_f16(a, bf, c);
      }
      int col = nt*16 + (lane & 15);
      float bb = bo[l*DIM + col];
      #pragma unroll
      for (int i = 0; i < 8; ++i) {
        int row = mt*16 + ((lane >> 4) << 3) + i;
        if (row < SEQ) X[row*DIM + col] += c[i] + bb;
      }
    }
    __syncthreads();

    // ---- LN2 -> A16 ----
    layernorm_rows(X, ln2_s + l*DIM, ln2_b + l*DIM, A16, wave, lane);
    __syncthreads();

    // ---- FFN, chunked over DFF (4 x 128 cols); down-proj accumulates in
    //      per-wave VGPR accumulators across chunks ----
    v8f acc[3];
    #pragma unroll
    for (int i = 0; i < 3; ++i) acc[i] = (v8f){0,0,0,0,0,0,0,0};

    for (int ch = 0; ch < 4; ++ch) {
      // g = gelu(h @ W1[:,chunk] + b1[chunk])  -> G16 (f16)
      const _Float16* W1c = W1Tl + (size_t)(ch*128) * DIM;
      for (int t = wave; t < 24; t += 8) {
        int mt = t >> 3, nt = t & 7;
        v8f c = {0,0,0,0,0,0,0,0};
        #pragma unroll
        for (int kk = 0; kk < 4; ++kk) {
          v16h a  = load_frag_lds(A16, mt*16, kk*32, DIM, lane);
          v16h bf = load_frag_g  (W1c, nt*16, kk*32, DIM, lane);
          c = wmma_f16(a, bf, c);
        }
        int col = nt*16 + (lane & 15);
        float b1v = b1[l*DFFN + ch*128 + col];
        #pragma unroll
        for (int i = 0; i < 8; ++i) {
          int row = mt*16 + ((lane >> 4) << 3) + i;
          float v = c[i] + b1v;
          v = 0.5f * v * (1.f + erff(v * 0.70710678118f)); // exact gelu
          G16[row*DIM + col] = (row < SEQ) ? (_Float16)v : (_Float16)0.f;
        }
      }
      __syncthreads();

      // acc += g @ W2[chunk,:]
      #pragma unroll
      for (int ti = 0; ti < 3; ++ti) {
        int t  = wave + ti*8;
        int mt = t >> 3, nt = t & 7;
        v8f c = acc[ti];
        #pragma unroll
        for (int kk = 0; kk < 4; ++kk) {
          v16h a  = load_frag_lds(G16,  mt*16, kk*32, DIM, lane);
          v16h bf = load_frag_g  (W2Tl, nt*16, ch*128 + kk*32, DFFN, lane);
          c = wmma_f16(a, bf, c);
        }
        acc[ti] = c;
      }
      __syncthreads();
    }

    // x += ffn + b2  (each (mt,nt) tile owned by exactly one wave)
    #pragma unroll
    for (int ti = 0; ti < 3; ++ti) {
      int t  = wave + ti*8;
      int mt = t >> 3, nt = t & 7;
      int col = nt*16 + (lane & 15);
      float b2v = b2[l*DIM + col];
      #pragma unroll
      for (int i = 0; i < 8; ++i) {
        int row = mt*16 + ((lane >> 4) << 3) + i;
        if (row < SEQ) X[row*DIM + col] += acc[ti][i] + b2v;
      }
    }
    __syncthreads();
  }

  // ---- final LN (row 0 only) + head + masking ----
  float* Yrow = SC;  // reuse scores region as f32 scratch
  if (wave == 0) {
    float vals[4]; float sum = 0.f, sum2 = 0.f;
    #pragma unroll
    for (int j = 0; j < 4; ++j) {
      float v = X[lane*4 + j]; vals[j] = v; sum += v; sum2 += v*v;
    }
    sum = wave_sum(sum); sum2 = wave_sum(sum2);
    float mu = sum * (1.f/DIM);
    float rinv = rsqrtf(sum2 * (1.f/DIM) - mu*mu + 1e-5f);
    #pragma unroll
    for (int j = 0; j < 4; ++j)
      Yrow[lane*4+j] = (vals[j]-mu)*rinv*lnf_s[lane*4+j] + lnf_b[lane*4+j];
  }
  __syncthreads();

  if (tid < CCOLS) {
    float acc = bhead[tid];
    for (int d = 0; d < DIM; ++d) acc += Yrow[d] * Whead[d*CCOLS + tid];
    int topcell = board[b*(RR*CCOLS) + tid];  // board[b, 0, tid]
    bool valid = (topcell == 0);
    out_logits[b*CCOLS + tid] = acc;
    out_masked[b*CCOLS + tid] = valid ? acc : -1e9f;
    out_valid [b*CCOLS + tid] = valid ? 1.f : 0.f;
  }
}

extern "C" void kernel_launch(void* const* d_in, const int* in_sizes, int n_in,
                              void* d_out, int out_size, void* d_ws, size_t ws_size,
                              hipStream_t stream) {
  const int*   board      = (const int*)  d_in[0];
  const int*   player     = (const int*)  d_in[1];
  const int*   bucket     = (const int*)  d_in[2];
  const float* cell_emb   = (const float*)d_in[3];
  const float* player_emb = (const float*)d_in[4];
  const float* cls_token  = (const float*)d_in[5];
  const float* bias_table = (const float*)d_in[6];
  const float* ln1_s      = (const float*)d_in[7];
  const float* ln1_b      = (const float*)d_in[8];
  const float* Wq         = (const float*)d_in[9];
  const float* bq         = (const float*)d_in[10];
  const float* Wk         = (const float*)d_in[11];
  const float* bk         = (const float*)d_in[12];
  const float* Wv         = (const float*)d_in[13];
  const float* bv         = (const float*)d_in[14];
  const float* Wo         = (const float*)d_in[15];
  const float* bo         = (const float*)d_in[16];
  const float* ln2_s      = (const float*)d_in[17];
  const float* ln2_b      = (const float*)d_in[18];
  const float* W1         = (const float*)d_in[19];
  const float* b1         = (const float*)d_in[20];
  const float* W2         = (const float*)d_in[21];
  const float* b2         = (const float*)d_in[22];
  const float* lnf_s      = (const float*)d_in[23];
  const float* lnf_b      = (const float*)d_in[24];
  const float* Whead      = (const float*)d_in[25];
  const float* bhead      = (const float*)d_in[26];

  const int B = in_sizes[0] / (RR * CCOLS);

  // Workspace: f16 transposed weights (~1.5 MB total).
  _Float16* wqT = (_Float16*)d_ws;
  _Float16* wkT = wqT + (size_t)NL*DIM*DIM;
  _Float16* wvT = wkT + (size_t)NL*DIM*DIM;
  _Float16* woT = wvT + (size_t)NL*DIM*DIM;
  _Float16* w1T = woT + (size_t)NL*DIM*DIM;
  _Float16* w2T = w1T + (size_t)NL*DFFN*DIM;

  const int nDD = NL*DIM*DIM, nDF = NL*DIM*DFFN;
  cvt_transpose_kernel<<<(nDD+255)/256, 256, 0, stream>>>(Wq, wqT, NL, DIM,  DIM);
  cvt_transpose_kernel<<<(nDD+255)/256, 256, 0, stream>>>(Wk, wkT, NL, DIM,  DIM);
  cvt_transpose_kernel<<<(nDD+255)/256, 256, 0, stream>>>(Wv, wvT, NL, DIM,  DIM);
  cvt_transpose_kernel<<<(nDD+255)/256, 256, 0, stream>>>(Wo, woT, NL, DIM,  DIM);
  cvt_transpose_kernel<<<(nDF+255)/256, 256, 0, stream>>>(W1, w1T, NL, DIM,  DFFN);
  cvt_transpose_kernel<<<(nDF+255)/256, 256, 0, stream>>>(W2, w2T, NL, DFFN, DIM);

  float* out_logits = (float*)d_out;
  float* out_masked = out_logits + (size_t)B * CCOLS;
  float* out_valid  = out_masked + (size_t)B * CCOLS;

  fused_transformer_fwd<<<B, 256, SMEM_BYTES, stream>>>(
      board, player, bucket, cell_emb, player_emb, cls_token, bias_table,
      ln1_s, ln1_b, bq, bk, bv, bo, ln2_s, ln2_b, b1, b2, lnf_s, lnf_b,
      Whead, bhead, wqT, wkT, wvT, woT, w1T, w2T,
      out_logits, out_masked, out_valid);
}
